// Attention_45286135169078
// MI455X (gfx1250) — matrix operand order repaired and checked by reading, not compiled, not run on hardware
//
#include <hip/hip_runtime.h>

typedef __attribute__((ext_vector_type(16))) __bf16 v16bf;
typedef __attribute__((ext_vector_type(8)))  __bf16 v8bf;
typedef __attribute__((ext_vector_type(8)))  float  v8f;

#define WMMA_BF16(a,b,c) __builtin_amdgcn_wmma_f32_16x16x32_bf16( \
    false,(a),false,(b),(short)0,(c),false,false)

constexpr int BATCH = 2, SEQ = 2048, EMB = 1024, NH = 16, HD = 64;
constexpr int MTOT  = BATCH * SEQ;   // 4096 rows

// Async raw copy: 16B global -> 16B LDS (per-lane addresses), ASYNCcnt-tracked.
__device__ __forceinline__ void async_copy_b128(unsigned int lds_off, const void* gaddr) {
  asm volatile("global_load_async_to_lds_b128 %0, %1, off"
               :: "v"(lds_off), "v"(gaddr) : "memory");
}
__device__ __forceinline__ void wait_asynccnt0() {
  asm volatile("s_wait_asynccnt 0" ::: "memory");
}
__device__ __forceinline__ unsigned int lds_off32(const void* p) {
  return (unsigned int)(unsigned long long)p;   // LDS aperture: low 32 bits = LDS offset
}

// ---------------------------------------------------------------------------
// Generic WMMA GEMM:  C[M x N] = A[M x 1024] * W[1024 x N] (+bias)
// MODE 0: QKV epilogue (scatter q/k/v bf16 to ws, k/v fp32 to present)
// MODE 1: projection epilogue (fp32 + bias to out)
// Block tile 128x64, 8 waves, each wave 2x2 WMMA 16x16 tiles, K-step 32.
// ---------------------------------------------------------------------------
template <typename AT, int MODE>
__global__ __launch_bounds__(256) void gemm_kernel(
    const AT* __restrict__ A, const float* __restrict__ W,
    const float* __restrict__ bias, int N,
    __bf16* __restrict__ q_ws, __bf16* __restrict__ k_ws,
    __bf16* __restrict__ v_ws, float* __restrict__ present,
    float* __restrict__ out)
{
  constexpr int BM = 128, BN = 64, BK = 32;
  constexpr int SA = 40;   // A LDS stride (bf16), 80B rows -> 16B aligned
  constexpr int SB = 48;   // B LDS stride (bf16), 96B rows -> 32B aligned
  __shared__ __align__(32) __bf16 As[BM * SA];
  __shared__ __align__(32) __bf16 Bs[BN * SB];

  const int tid  = threadIdx.x;
  const int lane = tid & 31;
  const int wave = tid >> 5;
  const int wm = wave >> 1, wn = wave & 1;
  const int bm = blockIdx.x * BM;
  const int bn = blockIdx.y * BN;
  const bool lolane = lane < 16;
  const int ln15 = lane & 15;
  const int abase = lolane ? 0 : 8;

  v8f acc[2][2] = {};
  constexpr int K = EMB;

  for (int kk = 0; kk < K; kk += BK) {
    __syncthreads();
    { // A tile 128x32 -> bf16 LDS, 16 contiguous elements per thread
      int id0 = tid * 16;
      int row = id0 >> 5;
      int col = id0 & 31;
      const AT* src = A + (size_t)(bm + row) * K + kk + col;
      __bf16* dst = As + row * SA + col;
#pragma unroll
      for (int e = 0; e < 16; ++e) dst[e] = (__bf16)src[e];
      if (kk + BK < K) __builtin_prefetch(src + BK, 0, 0);
    }
    { // B tile 32x64 -> transposed bf16 LDS Bs[n][k]
      int id0 = tid * 8;
      int kr = id0 >> 6;
      int nc = id0 & 63;
      const float* src = W + (size_t)(kk + kr) * N + bn + nc;
#pragma unroll
      for (int e = 0; e < 8; ++e) Bs[(nc + e) * SB + kr] = (__bf16)src[e];
      if (kk + BK < K) __builtin_prefetch(src + (size_t)BK * N, 0, 0);
    }
    __syncthreads();

    v16bf af[2], bfr[2];
#pragma unroll
    for (int t = 0; t < 2; ++t) {
      // A fragment: 16-bit A 16x32 layout (two 8-element runs per lane)
      int m = wm * 32 + t * 16 + ln15;
      v8bf l0 = *(const v8bf*)(As + m * SA + abase);
      v8bf l1 = *(const v8bf*)(As + m * SA + 16 + abase);
#pragma unroll
      for (int e = 0; e < 8; ++e) { af[t][e] = l0[e]; af[t][8 + e] = l1[e]; }
      // B fragment: same two 8-element K runs per lane as A
      int n = wn * 32 + t * 16 + ln15;
      v8bf b0 = *(const v8bf*)(Bs + n * SB + abase);
      v8bf b1 = *(const v8bf*)(Bs + n * SB + 16 + abase);
#pragma unroll
      for (int e = 0; e < 8; ++e) { bfr[t][e] = b0[e]; bfr[t][8 + e] = b1[e]; }
    }
#pragma unroll
    for (int i = 0; i < 2; ++i)
#pragma unroll
      for (int j = 0; j < 2; ++j)
        acc[i][j] = WMMA_BF16(af[i], bfr[j], acc[i][j]);
  }

  // Epilogue: C layout = element r -> row r + (lane<16?0:8), lane -> column
#pragma unroll
  for (int i = 0; i < 2; ++i) {
    int m0 = bm + wm * 32 + i * 16 + (lolane ? 0 : 8);
#pragma unroll
    for (int j = 0; j < 2; ++j) {
      int n = bn + wn * 32 + j * 16 + ln15;
      float bv = bias[n];
      if (MODE == 0) {
        int sec = n >> 10;          // 0:q 1:k 2:v
        int within = n & 1023;
        int h = within >> 6;
        int d = within & 63;
#pragma unroll
        for (int r = 0; r < 8; ++r) {
          int row = m0 + r;
          int b = row >> 11;        // /SEQ
          int s = row & 2047;
          float val = acc[i][j][r] + bv;
          size_t bh = (size_t)b * NH + h;
          if (sec == 0) {
            q_ws[(bh * SEQ + s) * HD + d] = (__bf16)val;
          } else if (sec == 1) {
            k_ws[(bh * SEQ + s) * HD + d] = (__bf16)val;
            present[((((size_t)0 * BATCH + b) * NH + h) * SEQ + s) * HD + d] = val;
          } else {
            v_ws[(bh * HD + d) * SEQ + s] = (__bf16)val;   // V stored transposed
            present[((((size_t)1 * BATCH + b) * NH + h) * SEQ + s) * HD + d] = val;
          }
        }
      } else {
#pragma unroll
        for (int r = 0; r < 8; ++r) {
          int row = m0 + r;
          out[(size_t)row * EMB + n] = acc[i][j][r] + bv;
        }
      }
    }
  }
}

// ---------------------------------------------------------------------------
// Causal flash attention. Block = 8 waves = 128 queries of one (b,h).
// K/V chunks (32 keys) are staged once per block into LDS via async copies;
// each wave runs online softmax on its 16-query tile.
// Q,K: [B*H][S][64] bf16; Vt: [B*H][64][S] bf16; Aout: [B][S][1024] bf16
// ---------------------------------------------------------------------------
__global__ __launch_bounds__(256) void attn_kernel(
    const __bf16* __restrict__ Q, const __bf16* __restrict__ Kw,
    const __bf16* __restrict__ Vt, __bf16* __restrict__ Aout)
{
  constexpr int KS = 80;   // Ks row stride (bf16): 160B, 32B aligned, bank-spread
  constexpr int VS = 48;   // Vs row stride (bf16):  96B, 32B aligned, bank-spread
  __shared__ __align__(32) __bf16 Ks[32 * KS];      // [key][hd]  5120B
  __shared__ __align__(32) __bf16 Vs[64 * VS];      // [hd][key]  6144B
  __shared__ __align__(32) __bf16 Pl[8][16 * 32];   // per-wave P transpose staging

  const int tid  = threadIdx.x;
  const int lane = tid & 31;
  const int wave = tid >> 5;
  const bool lolane = lane < 16;
  const int ln15 = lane & 15;
  const int abase = lolane ? 0 : 8;

  const int tiles_per_bh = SEQ / 128;                 // 16
  const int bh  = blockIdx.x / tiles_per_bh;
  const int q0b = (blockIdx.x % tiles_per_bh) * 128;  // block query start
  const int q0  = q0b + wave * 16;                    // wave query tile
  const int b = bh / NH, h = bh % NH;
  const __bf16* Qb = Q  + (size_t)bh * SEQ * HD;
  const __bf16* Kb = Kw + (size_t)bh * SEQ * HD;
  const __bf16* Vb = Vt + (size_t)bh * HD * SEQ;

  // Preload Q A-fragments for the two K=32 chunks of hd=64
  v16bf qf[2];
#pragma unroll
  for (int c = 0; c < 2; ++c) {
    const __bf16* qp = Qb + (size_t)(q0 + ln15) * HD + c * 32;
    v8bf l0 = *(const v8bf*)(qp + abase);
    v8bf l1 = *(const v8bf*)(qp + 16 + abase);
#pragma unroll
    for (int e = 0; e < 8; ++e) { qf[c][e] = l0[e]; qf[c][8 + e] = l1[e]; }
  }

  v8f o[4] = {};
  float mrun[8], lrun[8];
#pragma unroll
  for (int r = 0; r < 8; ++r) { mrun[r] = -3.0e38f; lrun[r] = 0.0f; }

  // Per-thread async copy addressing (16B granules)
  const int krow = tid >> 3, kcolb = (tid & 7) * 16;   // K chunk: 32 rows x 128B
  const int vrow = tid >> 2, vcolb = (tid & 3) * 16;   // V chunk: 64 rows x 64B
  const unsigned int ldsK = lds_off32(Ks) + krow * (KS * 2) + kcolb;
  const unsigned int ldsV = lds_off32(Vs) + vrow * (VS * 2) + vcolb;

  const int nchunks_blk = (q0b + 128) / 32;   // causal range for whole block
  for (int c0 = 0; c0 < nchunks_blk; ++c0) {
    const int k0 = c0 * 32;
    __syncthreads();   // previous chunk fully consumed
    // Stage K chunk [32 x 64] and V chunk [64 x 32] via async global->LDS
    async_copy_b128(ldsK, (const char*)Kb + (size_t)(k0 + krow) * (HD * 2) + kcolb);
    async_copy_b128(ldsV, (const char*)Vb + (size_t)vrow * (SEQ * 2) + k0 * 2 + vcolb);
    wait_asynccnt0();
    __syncthreads();   // chunk visible to all waves

    if (k0 >= q0 + 16) continue;   // fully masked for this wave's queries

    v8f s[2] = {};
#pragma unroll
    for (int ct = 0; ct < 2; ++ct) {
      int keyl = ct * 16 + ln15;
#pragma unroll
      for (int kc = 0; kc < 2; ++kc) {
        v16bf kf;
        {
          v8bf k0v = *(const v8bf*)(Ks + keyl * KS + kc * 32 + abase);
          v8bf k1v = *(const v8bf*)(Ks + keyl * KS + kc * 32 + 16 + abase);
#pragma unroll
          for (int e = 0; e < 8; ++e) { kf[e] = k0v[e]; kf[8 + e] = k1v[e]; }
        }
        s[ct] = WMMA_BF16(qf[kc], kf, s[ct]);
      }
      int key = k0 + keyl;
#pragma unroll
      for (int r = 0; r < 8; ++r) {   // scale 1/sqrt(64) + causal mask
        int qg = q0 + r + (lolane ? 0 : 8);
        float v = s[ct][r] * 0.125f;
        s[ct][r] = (key > qg) ? -1.0e30f : v;
      }
    }
    // Row max over 32 keys: element-wise then shfl over 16 lanes in row group
    float rm[8];
#pragma unroll
    for (int r = 0; r < 8; ++r) rm[r] = fmaxf(s[0][r], s[1][r]);
#pragma unroll
    for (int msk = 1; msk < 16; msk <<= 1)
#pragma unroll
      for (int r = 0; r < 8; ++r) rm[r] = fmaxf(rm[r], __shfl_xor(rm[r], msk, 32));

    float alpha[8], rs[8];
#pragma unroll
    for (int r = 0; r < 8; ++r) {
      float nm = fmaxf(mrun[r], rm[r]);
      alpha[r] = __expf(mrun[r] - nm);
      mrun[r] = nm;
      float p0 = __expf(s[0][r] - nm);
      float p1 = __expf(s[1][r] - nm);
      s[0][r] = p0; s[1][r] = p1;
      rs[r] = p0 + p1;
    }
#pragma unroll
    for (int msk = 1; msk < 16; msk <<= 1)
#pragma unroll
      for (int r = 0; r < 8; ++r) rs[r] += __shfl_xor(rs[r], msk, 32);
#pragma unroll
    for (int r = 0; r < 8; ++r) lrun[r] = lrun[r] * alpha[r] + rs[r];
#pragma unroll
    for (int t = 0; t < 4; ++t)
#pragma unroll
      for (int r = 0; r < 8; ++r) o[t][r] *= alpha[r];

    // C-layout -> A-layout transpose of P through per-wave LDS
    __bf16* pl = Pl[wave];
#pragma unroll
    for (int ct = 0; ct < 2; ++ct)
#pragma unroll
      for (int r = 0; r < 8; ++r)
        pl[(r + (lolane ? 0 : 8)) * 32 + ct * 16 + ln15] = (__bf16)s[ct][r];
    asm volatile("s_wait_dscnt 0" ::: "memory");   // same-wave DS ordering
    v16bf pf;
    {
      v8bf l0 = *(const v8bf*)(pl + ln15 * 32 + abase);
      v8bf l1 = *(const v8bf*)(pl + ln15 * 32 + 16 + abase);
#pragma unroll
      for (int e = 0; e < 8; ++e) { pf[e] = l0[e]; pf[8 + e] = l1[e]; }
    }
    // PV from LDS-staged transposed V: two 8-element key runs per lane (A order)
#pragma unroll
    for (int t = 0; t < 4; ++t) {
      int d = t * 16 + ln15;
      v16bf vf;
      v8bf v0 = *(const v8bf*)(Vs + d * VS + abase);
      v8bf v1 = *(const v8bf*)(Vs + d * VS + 16 + abase);
#pragma unroll
      for (int e = 0; e < 8; ++e) { vf[e] = v0[e]; vf[8 + e] = v1[e]; }
      o[t] = WMMA_BF16(pf, vf, o[t]);
    }
  }

  // Normalize and store merged-head activation (bf16) for projection
#pragma unroll
  for (int r = 0; r < 8; ++r) lrun[r] = 1.0f / lrun[r];
#pragma unroll
  for (int t = 0; t < 4; ++t) {
    int col = h * HD + t * 16 + ln15;
#pragma unroll
    for (int r = 0; r < 8; ++r) {
      int qg = q0 + r + (lolane ? 0 : 8);
      Aout[((size_t)(b * SEQ + qg)) * EMB + col] = (__bf16)(o[t][r] * lrun[r]);
    }
  }
}

// ---------------------------------------------------------------------------
extern "C" void kernel_launch(void* const* d_in, const int* in_sizes, int n_in,
                              void* d_out, int out_size, void* d_ws, size_t ws_size,
                              hipStream_t stream) {
  const float* x        = (const float*)d_in[0];
  const float* c_attn_w = (const float*)d_in[1];
  const float* c_attn_b = (const float*)d_in[2];
  const float* c_proj_w = (const float*)d_in[3];
  const float* c_proj_b = (const float*)d_in[4];

  float* out     = (float*)d_out;                       // a: [2,2048,1024]
  float* present = out + (size_t)MTOT * EMB;            // [2,2,16,2048,64]

  char* ws = (char*)d_ws;
  const size_t SEG = (size_t)8 * 1024 * 1024;           // 8 MB per segment
  __bf16* q_ws = (__bf16*)(ws);                         // [B,H,S,64]
  __bf16* k_ws = (__bf16*)(ws + SEG);                   // [B,H,S,64]
  __bf16* v_ws = (__bf16*)(ws + 2 * SEG);               // [B,H,64,S] transposed
  __bf16* a_ws = (__bf16*)(ws + 3 * SEG);               // [B,S,1024]

  // 1) QKV projection with scatter + present writeback
  dim3 g1(MTOT / 128, (3 * EMB) / 64);
  gemm_kernel<float, 0><<<g1, 256, 0, stream>>>(
      x, c_attn_w, c_attn_b, 3 * EMB, q_ws, k_ws, v_ws, present, nullptr);

  // 2) causal flash attention (block-cooperative K/V staging)
  attn_kernel<<<BATCH * NH * (SEQ / 128), 256, 0, stream>>>(q_ws, k_ws, v_ws, a_ws);

  // 3) output projection
  dim3 g3(MTOT / 128, EMB / 64);
  gemm_kernel<__bf16, 1><<<g3, 256, 0, stream>>>(
      a_ws, c_proj_w, c_proj_b, EMB, nullptr, nullptr, nullptr, nullptr, out);
}
